// ClusterPrediction_51805895524846
// MI455X (gfx1250) — matrix operand order repaired and checked by reading, not compiled, not run on hardware
//
#include <hip/hip_runtime.h>
#include <hip/hip_bf16.h>

typedef __attribute__((ext_vector_type(16))) _Float16 v16h;
typedef __attribute__((ext_vector_type(8)))  float    v8f;

#define NPTS  16384
#define NEDGE 262144
#define KNN   30
#define NS    0.2f
#define BUFSZ 160

__device__ __forceinline__ v8f wmma_f16(v16h a, v16h b, v8f c) {
  // v_wmma_f32_16x16x32_f16 : D = A(16x32 f16) x B(32x16 f16) + C(16x16 f32)
  return __builtin_amdgcn_wmma_f32_16x16x32_f16(false, a, false, b, (short)0, c,
                                                false, false);
}
__device__ __forceinline__ float lrelu(float z) { return z > 0.0f ? z : NS * z; }

// ---------------------------------------------------------------------------
// Weight staging: f32 -> f16, K padded to a multiple of 32 (zeros).
// ---------------------------------------------------------------------------
__global__ void prep_weights(const float* __restrict__ W1,
                             const float* __restrict__ W2,
                             const float* __restrict__ W3,
                             const float* __restrict__ Wh1,
                             _Float16* __restrict__ wh1,   // 32  x 64
                             _Float16* __restrict__ wh2,   // 128 x 64
                             _Float16* __restrict__ wh3,   // 128 x 64
                             _Float16* __restrict__ whh) { // 128 x 64
  int t = blockIdx.x * blockDim.x + threadIdx.x;
  if (t >= 128 * 64) return;
  int k = t >> 6, c = t & 63;
  wh2[t] = (_Float16)W2[t];
  wh3[t] = (_Float16)W3[t];
  whh[t] = (_Float16)Wh1[t];
  if (k < 32) wh1[t] = (k < 6) ? (_Float16)W1[k * 64 + c] : (_Float16)0.0f;
}

// ---------------------------------------------------------------------------
// Point staging: features -> f16 (zero-padded to 64) + squared norms.
// ---------------------------------------------------------------------------
__global__ void prep_points(const float* __restrict__ xin, int cin,
                            _Float16* __restrict__ xh, float* __restrict__ sqv) {
  int i = blockIdx.x * blockDim.x + threadIdx.x;
  if (i >= NPTS) return;
  float s = 0.0f;
  for (int c = 0; c < 64; ++c) {
    float v = (c < cin) ? xin[(size_t)i * cin + c] : 0.0f;
    _Float16 h = (_Float16)v;
    xh[(size_t)i * 64 + c] = h;
    float vf = (float)h;
    s += vf * vf;
  }
  sqv[i] = s;
}

// ---------------------------------------------------------------------------
// Fused Gram (WMMA) + top-30 selection. One block = 16 query rows, 8 waves
// each compute one 16x16 distance tile per 128-column iteration.
// Thresholds cached in registers (2x ds_load_b128 per iteration); squared
// row norms live in registers; next iteration's columns prefetched.
// ---------------------------------------------------------------------------
__global__ __launch_bounds__(256) void knn_kernel(const _Float16* __restrict__ xh,
                                                  const float* __restrict__ sqv,
                                                  int* __restrict__ knn_out) {
  __shared__ __align__(16) float topd[16][32];
  __shared__ int   topi[16][32];
  __shared__ __align__(16) float thr[16];
  __shared__ int   cnt[16];
  __shared__ float bufd[16][BUFSZ];
  __shared__ int   bufi[16][BUFSZ];

  const int tid  = threadIdx.x;
  const int lane = tid & 31;
  const int wave = tid >> 5;
  const int row0 = blockIdx.x * 16;

  if (tid < 16) {
    thr[tid] = 3.0e38f;
    cnt[tid] = 0;
    for (int s = 0; s < 32; ++s) { topd[tid][s] = 3.0e38f; topi[tid][s] = 0; }
  }
  __syncthreads();

  const int mrow = lane & 15, half = lane >> 4;

  // Squared norms of this tile's rows, kept in registers (m = r + 8*half).
  float rsqr[8];
#pragma unroll
  for (int r = 0; r < 8; ++r) rsqr[r] = sqv[row0 + 8 * half + r];

  // Pack A tile once: A[m][k] = xh[row0+m][k]  (16x64, two K=32 chunks)
  v16h a0, a1;
  {
    const _Float16* xp = xh + (size_t)(row0 + mrow) * 64;
#pragma unroll
    for (int s = 0; s < 16; ++s) {
      int ka = s + 8 * half + ((s >= 8) ? 8 : 0);
      a0[s] = xp[ka];
      a1[s] = xp[32 + ka];
    }
  }

  for (int it = 0; it < NPTS / 128; ++it) {
    int c0 = it * 128 + wave * 16;

    // Prefetch next iteration's rows for this wave (128B row, 64B per half).
    if (it + 1 < NPTS / 128) {
      const char* pf =
          (const char*)(xh + (size_t)(c0 + 128 + mrow) * 64) + half * 64;
      __builtin_prefetch(pf, 0, 1);
    }

    // Threshold snapshot for this iteration: thr[8*half .. 8*half+7].
    float tr[8];
    {
      const float4* tp = (const float4*)&thr[8 * half];
      float4 t0 = tp[0], t1 = tp[1];
      tr[0] = t0.x; tr[1] = t0.y; tr[2] = t0.z; tr[3] = t0.w;
      tr[4] = t1.x; tr[5] = t1.y; tr[6] = t1.z; tr[7] = t1.w;
    }

    // Pack B tile: B[k][n] = xh[c0+n][k]
    v16h b0, b1;
    {
      const _Float16* xq = xh + (size_t)(c0 + mrow) * 64;
#pragma unroll
      for (int s = 0; s < 16; ++s) {
        int kb = s + 8 * half + ((s >= 8) ? 8 : 0);
        b0[s] = xq[kb];
        b1[s] = xq[32 + kb];
      }
    }
    v8f acc = {};
    acc = wmma_f16(a0, b0, acc);
    acc = wmma_f16(a1, b1, acc);

    int col = c0 + mrow;            // D: N = lane&15
    float sqj = sqv[col];
#pragma unroll
    for (int r = 0; r < 8; ++r) {
      float d = rsqr[r] + sqj - 2.0f * acc[r];
      if (d < tr[r]) {
        int m = r + 8 * half;       // D: M = r + 8*half
        int pos = atomicAdd(&cnt[m], 1);
        if (pos < BUFSZ) { bufd[m][pos] = d; bufi[m][pos] = col; }
      }
    }
    __syncthreads();
    if (tid < 16) {
      int nc = cnt[tid];
      if (nc > 0) {
        if (nc > BUFSZ) nc = BUFSZ;
        for (int q = 0; q < nc; ++q) {
          float d = bufd[tid][q];
          if (d < topd[tid][KNN - 1]) {
            int ii = bufi[tid][q];
            int p = KNN - 1;
            while (p > 0 && topd[tid][p - 1] > d) {
              topd[tid][p] = topd[tid][p - 1];
              topi[tid][p] = topi[tid][p - 1];
              --p;
            }
            topd[tid][p] = d;
            topi[tid][p] = ii;
          }
        }
        cnt[tid] = 0;
        thr[tid] = topd[tid][KNN - 1];
      }
    }
    __syncthreads();
  }

  if (tid < 16) {
    for (int q = 0; q < KNN; ++q)
      knn_out[(size_t)(row0 + tid) * KNN + q] = topi[tid][q];
  }
}

// ---------------------------------------------------------------------------
// EdgeConv: per point, WMMA GEMM [xi, xj-xi] (30x2C, padded to 32 rows) @ W
// (2C x 64), then row-max (lrelu is monotonic -> applied after max).
// One wave per point, 8 points per block.
// ---------------------------------------------------------------------------
template <int CIN, int KC>
__global__ __launch_bounds__(256) void edge_conv_kernel(
    const float* __restrict__ xin, const int* __restrict__ nnidx,
    const _Float16* __restrict__ wh, const float* __restrict__ bias,
    float* __restrict__ xout) {
  __shared__ float xi_s[8][64];
  const int lane = threadIdx.x & 31;
  const int wave = threadIdx.x >> 5;
  const int pt = blockIdx.x * 8 + wave;

  for (int c = lane; c < 64; c += 32)
    xi_s[wave][c] = (c < CIN) ? xin[(size_t)pt * CIN + c] : 0.0f;
  __syncthreads();

  const int n = lane & 15, half = lane >> 4;
  const float* xi = xi_s[wave];
  v8f vmax[4];

#pragma unroll
  for (int mt = 0; mt < 2; ++mt) {
    int mr = mt * 16 + n;
    if (mr > KNN - 1) mr = KNN - 1;          // pad rows duplicate neighbor 29
    int j = nnidx[(size_t)pt * KNN + mr];
    const float* xj = xin + (size_t)j * CIN;

    v16h a[KC];
#pragma unroll
    for (int kc = 0; kc < KC; ++kc) {
#pragma unroll
      for (int s = 0; s < 16; ++s) {
        int k = kc * 32 + s + 8 * half + ((s >= 8) ? 8 : 0);
        float v = 0.0f;
        if (k < CIN) v = xi[k];
        else if (k < 2 * CIN) v = xj[k - CIN] - xi[k - CIN];
        a[kc][s] = (_Float16)v;
      }
    }
#pragma unroll
    for (int nt = 0; nt < 4; ++nt) {
      v8f acc = {};
#pragma unroll
      for (int kc = 0; kc < KC; ++kc) {
        v16h b;
#pragma unroll
        for (int s = 0; s < 16; ++s) {
          int kb = kc * 32 + s + 8 * half + ((s >= 8) ? 8 : 0);
          b[s] = wh[(size_t)kb * 64 + nt * 16 + n];
        }
        acc = wmma_f16(a[kc], b, acc);
      }
      if (mt == 0) vmax[nt] = acc;
      else {
#pragma unroll
        for (int r = 0; r < 8; ++r) vmax[nt][r] = fmaxf(vmax[nt][r], acc[r]);
      }
    }
  }

#pragma unroll
  for (int nt = 0; nt < 4; ++nt) {
    float mx = vmax[nt][0];
#pragma unroll
    for (int r = 1; r < 8; ++r) mx = fmaxf(mx, vmax[nt][r]);
    mx = fmaxf(mx, __shfl_xor(mx, 16, 32));  // merge row halves
    int ch = nt * 16 + n;
    if (half == 0) xout[(size_t)pt * 64 + ch] = lrelu(mx + bias[ch]);
  }
}

// ---------------------------------------------------------------------------
// Edge head: gather [x[v0], x[v1]] (E x 128), WMMA 128->64, lrelu, dot with
// Wh2 (butterfly reduce over the 16-lane half group), sigmoid.
// One wave per 16 edges, 8 waves per block.
// ---------------------------------------------------------------------------
__global__ __launch_bounds__(256) void edge_head_kernel(
    const float* __restrict__ x3, const int* __restrict__ ei,
    const _Float16* __restrict__ whh, const float* __restrict__ bh1,
    const float* __restrict__ wh2, const float* __restrict__ bh2,
    float* __restrict__ out) {
  const int lane = threadIdx.x & 31;
  const int wave = threadIdx.x >> 5;
  const int e0 = (blockIdx.x * 8 + wave) * 16;
  const int n = lane & 15, half = lane >> 4;

  int e = e0 + n;
  int p = ei[e], q = ei[NEDGE + e];
  int v0 = p < q ? p : q;
  int v1 = p < q ? q : p;
  const float* f0 = x3 + (size_t)v0 * 64;
  const float* f1 = x3 + (size_t)v1 * 64;

  v16h a[4];
#pragma unroll
  for (int kc = 0; kc < 4; ++kc) {
#pragma unroll
    for (int s = 0; s < 16; ++s) {
      int k = kc * 32 + s + 8 * half + ((s >= 8) ? 8 : 0);
      a[kc][s] = (_Float16)((k < 64) ? f0[k] : f1[k - 64]);
    }
  }

  float part[8];
#pragma unroll
  for (int r = 0; r < 8; ++r) part[r] = 0.0f;

#pragma unroll
  for (int nt = 0; nt < 4; ++nt) {
    v8f acc = {};
#pragma unroll
    for (int kc = 0; kc < 4; ++kc) {
      v16h b;
#pragma unroll
      for (int s = 0; s < 16; ++s) {
        int kb = kc * 32 + s + 8 * half + ((s >= 8) ? 8 : 0);
        b[s] = whh[(size_t)kb * 64 + nt * 16 + n];
      }
      acc = wmma_f16(a[kc], b, acc);
    }
    int ch = nt * 16 + n;
    float bb = bh1[ch], w2 = wh2[ch];
#pragma unroll
    for (int r = 0; r < 8; ++r) part[r] += lrelu(acc[r] + bb) * w2;
  }

#pragma unroll
  for (int r = 0; r < 8; ++r) {
    float t = part[r];
    t += __shfl_xor(t, 1, 32);
    t += __shfl_xor(t, 2, 32);
    t += __shfl_xor(t, 4, 32);
    t += __shfl_xor(t, 8, 32);
    part[r] = t;
  }
  if (n == 0) {
    float bb2 = bh2[0];
#pragma unroll
    for (int r = 0; r < 8; ++r) {
      float z = part[r] + bb2;
      out[e0 + r + 8 * half] = 1.0f / (1.0f + __expf(-z));
    }
  }
}

// ---------------------------------------------------------------------------
extern "C" void kernel_launch(void* const* d_in, const int* in_sizes, int n_in,
                              void* d_out, int out_size, void* d_ws, size_t ws_size,
                              hipStream_t stream) {
  (void)in_sizes; (void)n_in; (void)out_size; (void)ws_size;
  const float* x   = (const float*)d_in[0];
  const int*   ei  = (const int*)d_in[1];
  const float* W1  = (const float*)d_in[2];
  const float* b1  = (const float*)d_in[3];
  const float* W2  = (const float*)d_in[4];
  const float* b2  = (const float*)d_in[5];
  const float* W3  = (const float*)d_in[6];
  const float* b3  = (const float*)d_in[7];
  const float* Wh1 = (const float*)d_in[8];
  const float* bh1 = (const float*)d_in[9];
  const float* Wh2 = (const float*)d_in[10];
  const float* bh2 = (const float*)d_in[11];
  float* out = (float*)d_out;

  char* ws = (char*)d_ws;
  _Float16* xh  = (_Float16*)(ws + 0);         // 16384*64*2 = 2 MB
  float*    sqv = (float*)(ws + 2097152);      // 64 KB
  int*      idx = (int*)(ws + 2162688);        // 16384*30*4 ~ 1.9 MB
  float*    xa  = (float*)(ws + 4128768);      // 4 MB
  float*    xb  = (float*)(ws + 8323072);      // 4 MB
  _Float16* wh1 = (_Float16*)(ws + 12517376);  // 32*64*2
  _Float16* wh2 = (_Float16*)(ws + 12521472);  // 128*64*2
  _Float16* wh3 = (_Float16*)(ws + 12537856);  // 128*64*2
  _Float16* whh = (_Float16*)(ws + 12554240);  // 128*64*2

  prep_weights<<<32, 256, 0, stream>>>(W1, W2, W3, Wh1, wh1, wh2, wh3, whh);

  // layer 1 (C=3)
  prep_points<<<NPTS / 256, 256, 0, stream>>>(x, 3, xh, sqv);
  knn_kernel<<<NPTS / 16, 256, 0, stream>>>(xh, sqv, idx);
  edge_conv_kernel<3, 1><<<NPTS / 8, 256, 0, stream>>>(x, idx, wh1, b1, xa);

  // layer 2 (C=64)
  prep_points<<<NPTS / 256, 256, 0, stream>>>(xa, 64, xh, sqv);
  knn_kernel<<<NPTS / 16, 256, 0, stream>>>(xh, sqv, idx);
  edge_conv_kernel<64, 4><<<NPTS / 8, 256, 0, stream>>>(xa, idx, wh2, b2, xb);

  // layer 3 (C=64)
  prep_points<<<NPTS / 256, 256, 0, stream>>>(xb, 64, xh, sqv);
  knn_kernel<<<NPTS / 16, 256, 0, stream>>>(xh, sqv, idx);
  edge_conv_kernel<64, 4><<<NPTS / 8, 256, 0, stream>>>(xb, idx, wh3, b3, xa);

  // edge head
  edge_head_kernel<<<NEDGE / 128, 256, 0, stream>>>(xa, ei, whh, bh1, Wh2, bh2, out);
}
